// EdgeConvLayer_14886356648763
// MI455X (gfx1250) — compile-verified, matrix operand order checked by reading, not run on hardware
//
#include <hip/hip_runtime.h>

typedef __attribute__((ext_vector_type(16))) _Float16 v16h;
typedef __attribute__((ext_vector_type(8)))  _Float16 v8h;
typedef __attribute__((ext_vector_type(2)))  _Float16 v2h;
typedef __attribute__((ext_vector_type(8)))  float    v8f;

#define IN_F   64
#define OUT_F  64
#define EDGE_F 8
#define K1     136          // 2*IN_F + EDGE_F
#define K1P    160          // padded to multiple of 32
#define KC1    5            // layer-1 K chunks of 32
#define KC2    2            // layer-2 K chunks of 32
#define NWAVES 4
#define BLOCK  (NWAVES * 32)
#define TILES_PER_WAVE 8
#define GROUP_SPAN (NWAVES * TILES_PER_WAVE)   // tiles per block-group
#define MAX_BLOCKS 768
#define W1_ITERS ((K1 * OUT_F / 4) / BLOCK)    // 17
#define W2_ITERS ((OUT_F * OUT_F / 4) / BLOCK) // 8

// ---- fragment-layout helpers -----------------------------------------------
// Every 16x32 (A) / 32x16 (B) f16 WMMA operand lives in LDS pre-swizzled as
//   [frag][chunk(2)][lane(32)][8 halfs]
// so a lane fetches its v16h operand with two 16B ds_load_b128 (conflict-free:
// lane stride = 4 dwords -> 16 lanes cover all 64 banks exactly once).

__device__ __forceinline__ v16h loadFrag(const _Float16* base, int frag, int lane) {
    const v8h lo = *(const v8h*)(base + ((frag * 2 + 0) * 32 + lane) * 8);
    const v8h hi = *(const v8h*)(base + ((frag * 2 + 1) * 32 + lane) * 8);
    return __builtin_shufflevector(lo, hi,
        0, 1, 2, 3, 4, 5, 6, 7, 8, 9, 10, 11, 12, 13, 14, 15);
}

// Pointer to the row-0 slot of A column pair (c, c+1), c even.  Row r's slot
// is this + r*8 halfs (row bits sit below the K-half bit of the lane field).
__device__ __forceinline__ _Float16* aSlot(_Float16* base, int c) {
    int kc = c >> 5, c5 = c & 31;
    int j  = ((c5 >> 4) << 2) | ((c5 & 7) >> 1);
    int p  = j << 1;
    int laneHi = ((c5 >> 3) & 1) << 4;
    return base + ((kc * 2 + (p >> 3)) * 32 + laneHi) * 8 + (p & 7);
}

// B-matrix element (k, n) -> fragment (kc*4 + nt).
__device__ __forceinline__ void storeBElem(_Float16* base, int k, int n, float v) {
    int kc = k >> 5, nt = n >> 4;
    int c5 = k & 31;
    int j  = ((c5 >> 4) << 2) | ((c5 & 7) >> 1);
    int p  = (j << 1) | (c5 & 1);
    int lane = (n & 15) | (((c5 >> 3) & 1) << 4);
    base[(((kc * 4 + nt) * 2 + (p >> 3)) * 32 + lane) * 8 + (p & 7)] = (_Float16)v;
}

__global__ void edgeconv_zero_kernel(float* __restrict__ out, int n) {
    int i = blockIdx.x * blockDim.x + threadIdx.x;
    if (i < n) out[i] = 0.0f;
}

__global__ __launch_bounds__(BLOCK)
void edgeconv_wmma_kernel(const float* __restrict__ x,
                          const long long* __restrict__ ei,   // (2, E) int64
                          const float* __restrict__ ea,       // (E, 8)
                          const float* __restrict__ W1,       // (136, 64)
                          const float* __restrict__ b1,       // (64,)
                          const float* __restrict__ W2,       // (64, 64)
                          const float* __restrict__ b2,       // (64,)
                          float* __restrict__ out,            // (N, 64)
                          int E)
{
    __shared__ _Float16 sW1f[KC1 * 4 * 2 * 32 * 8];        // 20 frags = 20 KiB
    __shared__ _Float16 sW2f[KC2 * 4 * 2 * 32 * 8];        //  8 frags =  8 KiB
    __shared__ _Float16 sAf [NWAVES][KC1 * 2 * 32 * 8];    // 20 KiB
    __shared__ _Float16 sHf [NWAVES][KC2 * 2 * 32 * 8];    //  8 KiB

    const int tid  = threadIdx.x;
    const int lane = tid & 31;
    const int wave = tid >> 5;
    const int nrow = lane & 15;          // A-row / B-col / D-col for this lane
    const int hr   = (lane >> 4) << 3;   // C/D row base: 0 or 8

    // ---- stage weights into fragment layout (once per block) ----
    // 1) zero all W1 fragments with wide stores (also provides the K-pad).
    for (int i = tid; i < KC1 * 4 * 2 * 32; i += BLOCK) {
        v8h z = {};
        *(v8h*)(sW1f + i * 8) = z;
    }
    // 2) issue all weight loads up front (independent -> one latency).
    float4 w1v[W1_ITERS];
    #pragma unroll
    for (int it = 0; it < W1_ITERS; ++it)
        w1v[it] = *(const float4*)(W1 + (tid + it * BLOCK) * 4);
    float4 w2v[W2_ITERS];
    #pragma unroll
    for (int it = 0; it < W2_ITERS; ++it)
        w2v[it] = *(const float4*)(W2 + (tid + it * BLOCK) * 4);
    __syncthreads();   // zeros complete (loads still in flight behind it)
    // 3) swizzled stores.
    #pragma unroll
    for (int it = 0; it < W1_ITERS; ++it) {
        int i = tid + it * BLOCK;
        int k = i >> 4, n4 = (i & 15) << 2;
        storeBElem(sW1f, k, n4 + 0, w1v[it].x);
        storeBElem(sW1f, k, n4 + 1, w1v[it].y);
        storeBElem(sW1f, k, n4 + 2, w1v[it].z);
        storeBElem(sW1f, k, n4 + 3, w1v[it].w);
    }
    #pragma unroll
    for (int it = 0; it < W2_ITERS; ++it) {
        int i = tid + it * BLOCK;
        int k = i >> 4, n4 = (i & 15) << 2;
        storeBElem(sW2f, k, n4 + 0, w2v[it].x);
        storeBElem(sW2f, k, n4 + 1, w2v[it].y);
        storeBElem(sW2f, k, n4 + 2, w2v[it].z);
        storeBElem(sW2f, k, n4 + 3, w2v[it].w);
    }
    __syncthreads();

    // W2 fragments are small: keep them in registers for the whole kernel.
    v16h w2f[KC2 * 4];
    #pragma unroll
    for (int f = 0; f < KC2 * 4; ++f) w2f[f] = loadFrag(sW2f, f, lane);

    _Float16* sAfW = sAf[wave];
    _Float16* sHfW = sHf[wave];

    // Zero wave-private A fragments once (covers the K=136..159 padding).
    #pragma unroll
    for (int f = 0; f < KC1 * 2; ++f) {
        v8h z = {};
        *(v8h*)(sAfW + (f * 32 + lane) * 8) = z;
    }

    // Hoisted per-lane staging pointers (row-0 slots; row r = +r*8 halfs).
    _Float16* paSrc = aSlot(sAfW, 2 * lane);              // cols [0,64)
    _Float16* paDst = aSlot(sAfW, 64 + 2 * lane);         // cols [64,128)
    _Float16* paEa  = aSlot(sAfW, 128 + 2 * (lane & 3));  // cols [128,136)
    const bool eaLane = (lane < 4);

    // Hoisted h-store base: (nt,v) slot = +nt*256 + v*8.
    const int hBase = ((hr | ((nrow >> 3) << 4)) * 8) + (nrow & 7);

    // per-lane bias values for the 4 N-tiles (column = nt*16 + nrow)
    float b1v[4], b2v[4];
    #pragma unroll
    for (int nt = 0; nt < 4; ++nt) {
        b1v[nt] = b1[nt * 16 + nrow];
        b2v[nt] = b2[nt * 16 + nrow];
    }

    // Edge ids are int64 but values < N < 2^31: read only the low dwords.
    const int* srcLo = (const int*)ei;           // [2*e]
    const int* dstLo = (const int*)(ei + E);     // [2*e]

    const int tiles = (E + 15) >> 4;
    const int numGroups = (tiles + GROUP_SPAN - 1) / GROUP_SPAN;

    for (int g = blockIdx.x; g < numGroups; g += (int)gridDim.x) {
        int tileBase = g * GROUP_SPAN + wave * TILES_PER_WAVE;

        for (int t = 0; t < TILES_PER_WAVE; ++t) {
            int tile = tileBase + t;
            if (tile >= tiles) break;                 // wave-uniform
            int e0 = tile << 4;

            // ---- stage A tile: 16 edges x 136 live cols, f32->f16 ----
            for (int r = 0; r < 16; ++r) {
                int e = e0 + r;
                bool ok = (e < E);                    // wave-uniform
                int s = ok ? srcLo[e << 1] : 0;       // uniform scalar b32 loads
                int d = ok ? dstLo[e << 1] : 0;
                {
                    float2 v = ok ? *(const float2*)(x + s * IN_F + 2 * lane)
                                  : float2{0.f, 0.f};
                    v2h tp = { (_Float16)v.x, (_Float16)v.y };
                    *(v2h*)(paSrc + r * 8) = tp;
                }
                {
                    float2 v = ok ? *(const float2*)(x + d * IN_F + 2 * lane)
                                  : float2{0.f, 0.f};
                    v2h tp = { (_Float16)v.x, (_Float16)v.y };
                    *(v2h*)(paDst + r * 8) = tp;
                }
                if (eaLane) {
                    float2 v = ok ? *(const float2*)(ea + e * EDGE_F + 2 * lane)
                                  : float2{0.f, 0.f};
                    v2h tp = { (_Float16)v.x, (_Float16)v.y };
                    *(v2h*)(paEa + r * 8) = tp;
                }
            }

            // ---- layer 1: (16x160) @ (160x64), f16 in / f32 acc ----
            v8f acc[4] = {};
            #pragma unroll
            for (int kc = 0; kc < KC1; ++kc) {
                v16h a = loadFrag(sAfW, kc, lane);
                #pragma unroll
                for (int nt = 0; nt < 4; ++nt) {
                    v16h b = loadFrag(sW1f, kc * 4 + nt, lane);
                    acc[nt] = __builtin_amdgcn_wmma_f32_16x16x32_f16(
                        false, a, false, b, (short)0, acc[nt], false, false);
                }
            }

            // ---- bias + ReLU, write h into layer-2 A fragment layout ----
            #pragma unroll
            for (int nt = 0; nt < 4; ++nt) {
                #pragma unroll
                for (int v = 0; v < 8; ++v) {
                    float hv = acc[nt][v] + b1v[nt];
                    hv = hv > 0.0f ? hv : 0.0f;
                    sHfW[nt * 256 + hBase + v * 8] = (_Float16)hv;
                }
            }

            // ---- layer 2: (16x64) @ (64x64), W2 operands from registers ----
            v8f acc2[4] = {};
            #pragma unroll
            for (int kc = 0; kc < KC2; ++kc) {
                v16h a = loadFrag(sHfW, kc, lane);
                #pragma unroll
                for (int nt = 0; nt < 4; ++nt) {
                    acc2[nt] = __builtin_amdgcn_wmma_f32_16x16x32_f16(
                        false, a, false, w2f[kc * 4 + nt], (short)0, acc2[nt],
                        false, false);
                }
            }

            // ---- bias + atomic scatter-add to out[dst] ----
            #pragma unroll
            for (int v = 0; v < 8; ++v) {
                int e = e0 + hr + v;
                if (e < E) {
                    int dn = dstLo[e << 1];
                    float* orow = out + dn * OUT_F;
                    #pragma unroll
                    for (int nt = 0; nt < 4; ++nt) {
                        atomicAdd(orow + nt * 16 + nrow, acc2[nt][v] + b2v[nt]);
                    }
                }
            }
        }
    }
}

extern "C" void kernel_launch(void* const* d_in, const int* in_sizes, int n_in,
                              void* d_out, int out_size, void* d_ws, size_t ws_size,
                              hipStream_t stream) {
    (void)n_in; (void)d_ws; (void)ws_size;
    const float*     x   = (const float*)d_in[0];
    const long long* ei  = (const long long*)d_in[1];
    const float*     ea  = (const float*)d_in[2];
    const float*     W1  = (const float*)d_in[3];
    const float*     b1  = (const float*)d_in[4];
    const float*     W2  = (const float*)d_in[5];
    const float*     b2  = (const float*)d_in[6];
    float*           out = (float*)d_out;

    int E = in_sizes[1] / 2;

    edgeconv_zero_kernel<<<(out_size + 255) / 256, 256, 0, stream>>>(out, out_size);

    int tiles = (E + 15) >> 4;
    int numGroups = (tiles + GROUP_SPAN - 1) / GROUP_SPAN;
    int blocks = numGroups < MAX_BLOCKS ? numGroups : MAX_BLOCKS;
    if (blocks < 1) blocks = 1;
    edgeconv_wmma_kernel<<<blocks, BLOCK, 0, stream>>>(
        x, ei, ea, W1, b1, W2, b2, out, E);
}
